// TriangleMultiplication_88897233093052
// MI455X (gfx1250) — compile-verified, matrix operand order checked by reading, not run on hardware
//
#include <hip/hip_runtime.h>

// ---------------------------------------------------------------------------
// Triangle multiplication (outgoing) for MI455X / gfx1250.
// B=4, N=256, D=128. fp32 in/out, bf16 WMMA (v_wmma_f32_16x16x32_bf16) with
// f32 accumulation; all LDS staging via TDM (tensor_load_to_lds) synchronized
// with s_wait_tensorcnt + workgroup barriers.
// ---------------------------------------------------------------------------

typedef __attribute__((ext_vector_type(16))) __bf16        v16bf;
typedef __attribute__((ext_vector_type(8)))  float         v8f;
typedef __attribute__((ext_vector_type(4)))  __bf16        bf4;
typedef __attribute__((ext_vector_type(8)))  unsigned int  v8u;
typedef __attribute__((ext_vector_type(4)))  unsigned int  v4u;
typedef __attribute__((ext_vector_type(8)))  int           v8i;
typedef __attribute__((ext_vector_type(4)))  int           v4i;

#define DV 128
#define NN 256
#define BB 4

__device__ __forceinline__ float sigmoidf(float x) {
    return 1.0f / (1.0f + __expf(-x));
}

// The TDM writes LDS through descriptor data the compiler cannot see; without
// this barrier it treats the shared buffer as never-written (UB) and deletes
// the fragment loads (observed in round-2 disasm). Escaping the pointer into
// a memory-clobbering asm forces reloads.
__device__ __forceinline__ void lds_escape(const void* p) {
    asm volatile("" : : "r"(p) : "memory");
}

// CDNA5 16-bit WMMA fragment load (A-style or B-style) from a row-major
// matrix in LDS. Lane's row = (tile base + lane%16); per-lane data is two
// contiguous 16B chunks at byte offsets k0*2 + grp*16 and +32 (matches the
// ISA 16-bit A/B VGPR layout; lowers to two ds_load_b128).
__device__ __forceinline__ v16bf frag_ld(const __bf16* rowp, int k0, int grp) {
    const uint4* p = (const uint4*)((const char*)rowp + k0 * 2 + grp * 16);
    uint4 lo = p[0];
    uint4 hi = p[2];  // +32 bytes
    v8u u = {lo.x, lo.y, lo.z, lo.w, hi.x, hi.y, hi.z, hi.w};
    return __builtin_bit_cast(v16bf, u);
}

// ---------------------------------------------------------------------------
// TDM: async 2-D tile load Global -> LDS (D# per CDNA5 ISA ch.8).
// data_size = 2 bytes (bf16). Dims/strides in elements. Issued per-wave.
// ---------------------------------------------------------------------------
__device__ __forceinline__ void tdm_load_2d(unsigned lds_byte_off,
                                            const __bf16* gaddr,
                                            unsigned tensor_d0, unsigned tensor_d1,
                                            unsigned tile_d0, unsigned tile_d1,
                                            unsigned stride0) {
    unsigned long long ga = (unsigned long long)gaddr;
    v4u g0;
    g0[0] = 1u;                                             // count=1, user desc
    g0[1] = lds_byte_off;                                   // lds_addr
    g0[2] = (unsigned)(ga & 0xffffffffu);                   // global_addr[31:0]
    g0[3] = (unsigned)((ga >> 32) & 0x01ffffffu)            // global_addr[56:32]
            | 0x80000000u;                                  // type=2 ("image")
    v8i g1;
    g1[0] = (int)(1u << 16);                                // data_size=1 -> 2B
    g1[1] = (int)((tensor_d0 & 0xffffu) << 16);             // tensor_dim0[15:0]
    g1[2] = (int)(((tensor_d0 >> 16) & 0xffffu) |
                  ((tensor_d1 & 0xffffu) << 16));           // dim0 hi | dim1 lo
    g1[3] = (int)(((tensor_d1 >> 16) & 0xffffu) |
                  ((tile_d0 & 0xffffu) << 16));             // dim1 hi | tile0
    g1[4] = (int)(tile_d1 & 0xffffu);                       // tile1 (tile2=0)
    g1[5] = (int)stride0;                                   // dim0_stride[31:0]
    g1[6] = 0;                                              // stride hi bits
    g1[7] = 0;
    v4i z4 = {0, 0, 0, 0};
#if __clang_major__ >= 23
    v8i z8 = {0, 0, 0, 0, 0, 0, 0, 0};
    __builtin_amdgcn_tensor_load_to_lds(g0, g1, z4, z4, z8, 0);
#else
    __builtin_amdgcn_tensor_load_to_lds(g0, g1, z4, z4, 0);
#endif
}

// ---------------------------------------------------------------------------
// f32 -> bf16 conversion (weights, once).
// ---------------------------------------------------------------------------
__global__ __launch_bounds__(256) void k_cvt_bf16(const float* __restrict__ src,
                                                  __bf16* __restrict__ dst, int n) {
    int e = (blockIdx.x * 256 + threadIdx.x) * 4;
    if (e < n) {
        float4 w = *(const float4*)(src + e);
        bf4 o = {(__bf16)w.x, (__bf16)w.y, (__bf16)w.z, (__bf16)w.w};
        *(bf4*)(dst + e) = o;
    }
}

// ---------------------------------------------------------------------------
// LayerNorm over D=128: f32 in -> bf16 out. One wave32 per row.
// ---------------------------------------------------------------------------
__global__ __launch_bounds__(256) void k_layernorm_bf16(
    const float* __restrict__ x, const float* __restrict__ g,
    const float* __restrict__ b, __bf16* __restrict__ y, int nrows) {
    int lane = threadIdx.x & 31;
    int wave = threadIdx.x >> 5;
    long row = (long)blockIdx.x * 8 + wave;
    if (row >= nrows) return;

    float4 xv = ((const float4*)(x + row * DV))[lane];
    float s  = xv.x + xv.y + xv.z + xv.w;
    float ss = xv.x * xv.x + xv.y * xv.y + xv.z * xv.z + xv.w * xv.w;
#pragma unroll
    for (int off = 16; off > 0; off >>= 1) {
        s  += __shfl_xor(s,  off, 32);
        ss += __shfl_xor(ss, off, 32);
    }
    float mean = s * (1.0f / DV);
    float var  = ss * (1.0f / DV) - mean * mean;
    float rstd = rsqrtf(var + 1e-5f);

    float4 gv = ((const float4*)g)[lane];
    float4 bv = ((const float4*)b)[lane];
    bf4 o;
    o[0] = (__bf16)((xv.x - mean) * rstd * gv.x + bv.x);
    o[1] = (__bf16)((xv.y - mean) * rstd * gv.y + bv.y);
    o[2] = (__bf16)((xv.z - mean) * rstd * gv.z + bv.z);
    o[3] = (__bf16)((xv.w - mean) * rstd * gv.w + bv.w);
    *(bf4*)(y + row * DV + lane * 4) = o;
}

// ---------------------------------------------------------------------------
// WMMA GEMM, X[M,128] x W^T (torch Linear), MODE compile-time:
//  0: out_bf16 = sigmoid(X@Wg^T+bg) * (X@Wp^T+bp)
//  1: out_bf16 = sigmoid(X@Wg^T+bg)
//  2: out_f32  = (X@Wp^T+bp) * gate[m,o]
// 256 threads (8 waves); tile 128 rows x 128 cols; K=128 in 4 WMMA steps.
// X tile and weights staged to LDS by TDM (wave 0 issues the descriptors).
// ---------------------------------------------------------------------------
template <int MODE>
__global__ __launch_bounds__(256) void k_gemm_gated(
    const __bf16* __restrict__ X,
    const __bf16* __restrict__ Wgb, const float* __restrict__ bgf,
    const __bf16* __restrict__ Wpb, const float* __restrict__ bpf,
    const __bf16* __restrict__ gate,
    __bf16* __restrict__ obf, float* __restrict__ of32) {
    // [0]=X (32 KB), [1]=Wg (32 KB), [2]=Wp (32 KB)
    __shared__ __bf16 sm[3 * 128 * DV];
    __bf16* Xs  = sm;
    __bf16* Wgs = sm + 128 * DV;
    __bf16* Wps = sm + 2 * 128 * DV;

    constexpr bool NG = (MODE != 2);
    constexpr bool NP = (MODE != 1);

    int  tid = threadIdx.x;
    long m0  = (long)blockIdx.x * 128;

    if (tid < 32) {  // wave 0 issues DMA; TENSORcnt is per-wave
        tdm_load_2d(0u, X + m0 * DV, DV, 1u << 20, 128, 128, DV);
        if (NG) tdm_load_2d(32768u, Wgb, DV, DV, 128, 128, DV);
        if (NP) tdm_load_2d(65536u, Wpb, DV, DV, 128, 128, DV);
        __builtin_amdgcn_s_wait_tensorcnt(0);
    }
    __syncthreads();
    lds_escape(sm);

    int lane = tid & 31, wave = tid >> 5;
    int grp = lane >> 4, lr = lane & 15;
    const __bf16* arow = Xs + (wave * 16 + lr) * DV;

#pragma unroll
    for (int half = 0; half < 2; ++half) {
        v8f accg[4], accp[4];
#pragma unroll
        for (int n = 0; n < 4; ++n) {
            accg[n] = (v8f){0.f, 0.f, 0.f, 0.f, 0.f, 0.f, 0.f, 0.f};
            accp[n] = (v8f){0.f, 0.f, 0.f, 0.f, 0.f, 0.f, 0.f, 0.f};
        }
#pragma unroll
        for (int ks = 0; ks < 4; ++ks) {
            v16bf af = frag_ld(arow, ks * 32, grp);
#pragma unroll
            for (int n = 0; n < 4; ++n) {
                int colr = (half * 4 + n) * 16 + lr;
                if (NG) {
                    v16bf wg = frag_ld(Wgs + colr * DV, ks * 32, grp);
                    accg[n] = __builtin_amdgcn_wmma_f32_16x16x32_bf16(
                        false, af, false, wg, (short)0, accg[n], false, false);
                }
                if (NP) {
                    v16bf wp = frag_ld(Wps + colr * DV, ks * 32, grp);
                    accp[n] = __builtin_amdgcn_wmma_f32_16x16x32_bf16(
                        false, af, false, wp, (short)0, accp[n], false, false);
                }
            }
        }
#pragma unroll
        for (int n = 0; n < 4; ++n) {
            int col  = (half * 4 + n) * 16 + lr;
            float bgv = NG ? bgf[col] : 0.0f;
            float bpv = NP ? bpf[col] : 0.0f;
#pragma unroll
            for (int v = 0; v < 8; ++v) {
                long row = m0 + wave * 16 + v + 8 * grp;
                long idx = row * DV + col;
                if (MODE == 0) {
                    float gg = sigmoidf(accg[n][v] + bgv);
                    float pp = accp[n][v] + bpv;
                    obf[idx] = (__bf16)(gg * pp);
                } else if (MODE == 1) {
                    obf[idx] = (__bf16)sigmoidf(accg[n][v] + bgv);
                } else {
                    float pp = accp[n][v] + bpv;
                    of32[idx] = pp * (float)gate[idx];
                }
            }
        }
    }
}

// ---------------------------------------------------------------------------
// bf16 transpose: [B, N(r), N(c), D] -> [B, D, N(r), N(c)].
// ---------------------------------------------------------------------------
__global__ void k_tr_bf16(const __bf16* __restrict__ in, __bf16* __restrict__ out) {
    __shared__ __bf16 tile[32][33];
    int b  = blockIdx.z;
    int r  = blockIdx.y;
    int ct = blockIdx.x >> 2;  // N/32 = 8
    int dt = blockIdx.x & 3;   // D/32 = 4
    int tx = threadIdx.x, ty = threadIdx.y;
#pragma unroll
    for (int yy = 0; yy < 4; ++yy) {
        int c = ct * 32 + ty + yy * 8;
        int d = dt * 32 + tx;
        tile[ty + yy * 8][tx] = in[(((long)b * NN + r) * NN + c) * DV + d];
    }
    __syncthreads();
#pragma unroll
    for (int yy = 0; yy < 4; ++yy) {
        int d = dt * 32 + ty + yy * 8;
        int c = ct * 32 + tx;
        out[(((long)b * DV + d) * NN + r) * NN + c] = tile[tx][ty + yy * 8];
    }
}

// ---------------------------------------------------------------------------
// f32 transpose: [B, D, N(i), N(j)] -> [B, N(i), N(j), D].
// ---------------------------------------------------------------------------
__global__ void k_tr_f32(const float* __restrict__ in, float* __restrict__ out) {
    __shared__ float tile[32][33];
    int b  = blockIdx.z;
    int i  = blockIdx.y;
    int jt = blockIdx.x >> 2;
    int dt = blockIdx.x & 3;
    int tx = threadIdx.x, ty = threadIdx.y;
#pragma unroll
    for (int yy = 0; yy < 4; ++yy) {
        int d = dt * 32 + ty + yy * 8;
        int j = jt * 32 + tx;
        tile[ty + yy * 8][tx] = in[(((long)b * DV + d) * NN + i) * NN + j];
    }
    __syncthreads();
#pragma unroll
    for (int yy = 0; yy < 4; ++yy) {
        int j = jt * 32 + ty + yy * 8;
        int d = dt * 32 + tx;
        out[(((long)b * NN + i) * NN + j) * DV + d] = tile[tx][ty + yy * 8];
    }
}

// ---------------------------------------------------------------------------
// Triangle einsum: per slice s=(b,d), C[i,j] = sum_k A[i,k]*Bm[j,k].
// Slices are contiguous 256x256 bf16. 128x128 C-tile per block.
// LDS staging via TDM, double-buffered across the two 128-wide K chunks;
// waits via s_wait_tensorcnt + barrier + lds_escape. 128 KB LDS.
// ---------------------------------------------------------------------------
__global__ __launch_bounds__(256) void k_tri_gemm(
    const __bf16* __restrict__ aT, const __bf16* __restrict__ bT,
    float* __restrict__ outT) {
    // [0]=A kk0, [1]=B kk0, [2]=A kk1, [3]=B kk1 ; each 128x128 bf16 = 32 KB.
    __shared__ __bf16 smem[4 * 128 * 128];

    int  s  = blockIdx.z;
    int  i0 = blockIdx.y * 128;
    int  j0 = blockIdx.x * 128;
    long slice = (long)s * NN * NN;
    int  tid = threadIdx.x;
    int  lane = tid & 31, wave = tid >> 5;
    int  grp = lane >> 4, lr = lane & 15;

    const __bf16* gA0 = aT + slice + (long)i0 * NN;        // kk=0
    const __bf16* gB0 = bT + slice + (long)j0 * NN;
    const __bf16* gA1 = gA0 + 128;                         // kk=1 (K offset)
    const __bf16* gB1 = gB0 + 128;

    // Kick off kk=0 tiles (wave 0 issues; TENSORcnt is per-wave).
    if (tid < 32) {
        tdm_load_2d(0u,      gA0, NN, NN, 128, 128, NN);
        tdm_load_2d(32768u,  gB0, NN, NN, 128, 128, NN);
        __builtin_amdgcn_s_wait_tensorcnt(0);
    }
    __syncthreads();
    lds_escape(smem);

    // Prefetch kk=1 tiles while computing kk=0.
    if (tid < 32) {
        tdm_load_2d(65536u,  gA1, NN, NN, 128, 128, NN);
        tdm_load_2d(98304u,  gB1, NN, NN, 128, 128, NN);
    }

    v8f acc[8];
#pragma unroll
    for (int n = 0; n < 8; ++n)
        acc[n] = (v8f){0.f, 0.f, 0.f, 0.f, 0.f, 0.f, 0.f, 0.f};

#pragma unroll
    for (int kk = 0; kk < 2; ++kk) {
        const __bf16* As = smem + (kk * 2 + 0) * (128 * 128);
        const __bf16* Bs = smem + (kk * 2 + 1) * (128 * 128);
        const __bf16* arow = As + (wave * 16 + lr) * 128;
#pragma unroll
        for (int ks = 0; ks < 4; ++ks) {
            v16bf af = frag_ld(arow, ks * 32, grp);
#pragma unroll
            for (int n = 0; n < 8; ++n) {
                v16bf bf_ = frag_ld(Bs + (n * 16 + lr) * 128, ks * 32, grp);
                acc[n] = __builtin_amdgcn_wmma_f32_16x16x32_bf16(
                    false, af, false, bf_, (short)0, acc[n], false, false);
            }
        }
        if (kk == 0) {  // release kk=1 buffers once DMA completes
            if (tid < 32) __builtin_amdgcn_s_wait_tensorcnt(0);
            __syncthreads();
            lds_escape(smem);
        }
    }

#pragma unroll
    for (int n = 0; n < 8; ++n) {
        int col = j0 + n * 16 + lr;
#pragma unroll
        for (int v = 0; v < 8; ++v) {
            int row = i0 + wave * 16 + v + 8 * grp;
            outT[slice + (long)row * NN + col] = acc[n][v];
        }
    }
}

// ---------------------------------------------------------------------------
// Host-side orchestration (all launches on `stream`, graph-capture safe).
// ---------------------------------------------------------------------------
extern "C" void kernel_launch(void* const* d_in, const int* in_sizes, int n_in,
                              void* d_out, int out_size, void* d_ws, size_t ws_size,
                              hipStream_t stream) {
    (void)in_sizes; (void)n_in; (void)out_size; (void)ws_size;

    const float* z     = (const float*)d_in[0];
    const float* g_in  = (const float*)d_in[1];
    const float* b_in  = (const float*)d_in[2];
    const float* wpa   = (const float*)d_in[3];
    const float* bpa   = (const float*)d_in[4];
    const float* wga   = (const float*)d_in[5];
    const float* bga   = (const float*)d_in[6];
    const float* wpb   = (const float*)d_in[7];
    const float* bpb   = (const float*)d_in[8];
    const float* wgb   = (const float*)d_in[9];
    const float* bgb   = (const float*)d_in[10];
    const float* g_out = (const float*)d_in[11];
    const float* b_out = (const float*)d_in[12];
    const float* wop   = (const float*)d_in[13];
    const float* bop   = (const float*)d_in[14];
    const float* wog   = (const float*)d_in[15];
    const float* bog   = (const float*)d_in[16];

    const long   M     = (long)BB * NN * NN;   // 262144 rows
    const size_t SZ_BF = (size_t)M * DV * 2;   // 64 MB
    const size_t SZ_F  = (size_t)M * DV * 4;   // 128 MB

    char*   ws    = (char*)d_ws;
    __bf16* zl    = (__bf16*)(ws);                 // later reused as aT
    __bf16* abuf  = (__bf16*)(ws + SZ_BF);         // later reused as bT
    __bf16* bbuf  = (__bf16*)(ws + 2 * SZ_BF);     // later reused as lnout
    __bf16* og    = (__bf16*)(ws + 3 * SZ_BF);
    float*  outT  = (float*)(ws + 4 * SZ_BF);
    float*  outRM = (float*)(ws + 4 * SZ_BF + SZ_F);
    __bf16* wbase = (__bf16*)(ws + 4 * SZ_BF + 2 * SZ_F);
    __bf16* wga_b = wbase + 0 * DV * DV;
    __bf16* wpa_b = wbase + 1 * DV * DV;
    __bf16* wgb_b = wbase + 2 * DV * DV;
    __bf16* wpb_b = wbase + 3 * DV * DV;
    __bf16* wog_b = wbase + 4 * DV * DV;
    __bf16* wop_b = wbase + 5 * DV * DV;

    dim3 b256(256);
    dim3 trb(32, 8);
    dim3 trg(32, 256, 4);
    const int NW = DV * DV;  // 16384 elems per weight matrix

    // 0) weights -> bf16 (once)
    k_cvt_bf16<<<16, b256, 0, stream>>>(wga, wga_b, NW);
    k_cvt_bf16<<<16, b256, 0, stream>>>(wpa, wpa_b, NW);
    k_cvt_bf16<<<16, b256, 0, stream>>>(wgb, wgb_b, NW);
    k_cvt_bf16<<<16, b256, 0, stream>>>(wpb, wpb_b, NW);
    k_cvt_bf16<<<16, b256, 0, stream>>>(wog, wog_b, NW);
    k_cvt_bf16<<<16, b256, 0, stream>>>(wop, wop_b, NW);
    // 1) zl = LN(z)
    k_layernorm_bf16<<<(int)(M / 8), b256, 0, stream>>>(z, g_in, b_in, zl, (int)M);
    // 2) og = sigmoid(zl @ wog^T + bog)
    k_gemm_gated<1><<<(int)(M / 128), b256, 0, stream>>>(zl, wog_b, bog, wog_b, bog,
                                                         nullptr, og, nullptr);
    // 3) a = sigmoid(zl@wga^T+bga) * (zl@wpa^T+bpa)
    k_gemm_gated<0><<<(int)(M / 128), b256, 0, stream>>>(zl, wga_b, bga, wpa_b, bpa,
                                                         nullptr, abuf, nullptr);
    // 4) b = sigmoid(zl@wgb^T+bgb) * (zl@wpb^T+bpb)
    k_gemm_gated<0><<<(int)(M / 128), b256, 0, stream>>>(zl, wgb_b, bgb, wpb_b, bpb,
                                                         nullptr, bbuf, nullptr);
    // 5) transpose to per-channel slices
    __bf16* aT = zl;
    k_tr_bf16<<<trg, trb, 0, stream>>>(abuf, aT);
    __bf16* bT = abuf;
    k_tr_bf16<<<trg, trb, 0, stream>>>(bbuf, bT);
    // 6) triangle GEMMs (TDM-staged): outT[b,d,i,j] = sum_k aT[i,k]*bT[j,k]
    k_tri_gemm<<<dim3(2, 2, BB * DV), b256, 0, stream>>>(aT, bT, outT);
    // 7) back to row-major channels
    k_tr_f32<<<trg, trb, 0, stream>>>(outT, outRM);
    // 8) LN(out)
    __bf16* lnout = bbuf;
    k_layernorm_bf16<<<(int)(M / 8), b256, 0, stream>>>(outRM, g_out, b_out, lnout, (int)M);
    // 9) final: d_out = og * (lnout @ wop^T + bop)
    k_gemm_gated<2><<<(int)(M / 128), b256, 0, stream>>>(lnout, wop_b, bop, wop_b, bop,
                                                         og, nullptr, (float*)d_out);
}
